// GATPyG_6064493821982
// MI455X (gfx1250) — matrix operand order repaired and checked by reading, not compiled, not run on hardware
//
#include <hip/hip_runtime.h>
#include <hip/hip_bf16.h>
#include <string.h>

// Problem constants (from the reference)
#define NN   50000
#define EE   800000
#define DIN  64
#define DH   16
#define HEADS 8
#define C1   (HEADS * DH)   // 128
#define DOUT 16

typedef __attribute__((ext_vector_type(16))) _Float16 v16h;
typedef __attribute__((ext_vector_type(8)))  float    v8f;

union HVec { v16h v; _Float16 h[16]; };
union FVec { v8f  v; float    f[8];  };

// ---------- ordered-uint encoding for float atomic max ----------
__device__ __forceinline__ unsigned ordEnc(float f) {
    unsigned u = __float_as_uint(f);
    return (u & 0x80000000u) ? ~u : (u | 0x80000000u);
}
__device__ __forceinline__ float ordDec(unsigned e) {
    unsigned u = (e & 0x80000000u) ? (e & 0x7FFFFFFFu) : ~e;
    return __uint_as_float(u);
}

// ---------- WMMA fragment loaders (fp32 memory -> f16 fragments) ----------
// A: 16x32 (MxK), row-major source with leading dim lda.
// lanes 0-15: row = lane, K = kb + {0..7, 16..23}; lanes 16-31: K = kb + {8..15, 24..31}
__device__ __forceinline__ v16h load_a_frag(const float* __restrict__ A, int lda,
                                            int mBase, int kb, int lane) {
    HVec a;
    const int row   = mBase + (lane & 15);
    const int kbase = kb + ((lane >> 4) << 3);
    const float* ap = A + (size_t)row * lda + kbase;
#pragma unroll
    for (int j = 0; j < 16; ++j) {
        int k = j + ((j >= 8) ? 8 : 0);   // two contiguous 8-float runs -> b128 loads
        a.h[j] = (_Float16)ap[k];
    }
    return a.v;
}
// B: 32x16 (KxN), row-major source with leading dim ldb.
// lanes 0-15: col = lane, K = kb + {0..7, 16..23}; lanes 16-31: K = kb + {8..15, 24..31}
__device__ __forceinline__ v16h load_b_frag(const float* __restrict__ B, int ldb,
                                            int kb, int nBase, int lane) {
    HVec b;
    const int n     = nBase + (lane & 15);
    const int kbase = kb + ((lane >> 4) << 3);
    const float* bp = B + (size_t)kbase * ldb + n;
#pragma unroll
    for (int j = 0; j < 16; ++j)
        b.h[j] = (_Float16)bp[(size_t)(j + ((j >= 8) ? 8 : 0)) * ldb];
    return b.v;
}

// ---------- fused dual-output WMMA GEMM ----------
// Cl = A @ Bl, Cr = A @ Br with A[M, KB*32], B[KB*32, NT*16].
// One wave owns a 16-row strip: A fragments loaded ONCE into registers and
// reused across all NT column tiles of BOTH weight matrices.
// 4 waves per workgroup; whole-wave early-out keeps EXEC all-1s at WMMA.
template<int KB, int NT>
__global__ void wmma_dual_gemm(const float* __restrict__ A,
                               const float* __restrict__ Bl, const float* __restrict__ Br,
                               float* __restrict__ Cl, float* __restrict__ Cr, int M) {
    constexpr int K  = KB * 32;
    constexpr int Nc = NT * 16;
    const int lane = threadIdx.x & 31;
    const int warp = threadIdx.x >> 5;
    const int tile = blockIdx.x * 4 + warp;
    if (tile * 16 >= M) return;               // uniform per-wave: EXEC stays full
    const int mBase = tile << 4;

    v16h af[KB];
#pragma unroll
    for (int kb = 0; kb < KB; ++kb) {
        __builtin_prefetch(A + (size_t)(mBase + 16 + (lane & 15)) * K, 0, 1);
        af[kb] = load_a_frag(A, K, mBase, kb * 32, lane);
    }

    const int n    = lane & 15;
    const int mOff = mBase + ((lane >> 4) << 3);
    for (int nt = 0; nt < NT; ++nt) {
        FVec accL, accR;
#pragma unroll
        for (int r = 0; r < 8; ++r) { accL.f[r] = 0.0f; accR.f[r] = 0.0f; }
#pragma unroll
        for (int kb = 0; kb < KB; ++kb) {
            v16h bL = load_b_frag(Bl, Nc, kb * 32, nt * 16, lane);
            accL.v = __builtin_amdgcn_wmma_f32_16x16x32_f16(false, af[kb], false, bL,
                                                            (short)0, accL.v, false, false);
            v16h bR = load_b_frag(Br, Nc, kb * 32, nt * 16, lane);
            accR.v = __builtin_amdgcn_wmma_f32_16x16x32_f16(false, af[kb], false, bR,
                                                            (short)0, accR.v, false, false);
        }
        const int col = nt * 16 + n;
#pragma unroll
        for (int r = 0; r < 8; ++r) {
            Cl[(size_t)(mOff + r) * Nc + col] = accL.f[r];
            Cr[(size_t)(mOff + r) * Nc + col] = accR.f[r];
        }
    }
}

// ---------- fused edge kernel: e = ef@W (WMMA), logits + atomic seg-max ----------
// One wave handles a tile of 16 edges. K is fixed at 64 (edge_feats width).
__global__ void edge_attn_logits(const float* __restrict__ ef, const float* __restrict__ W,
                                 const float* __restrict__ XL, const float* __restrict__ XR,
                                 const float* __restrict__ att,
                                 const int* __restrict__ src, const int* __restrict__ dst,
                                 float* __restrict__ logits, unsigned* __restrict__ maxU,
                                 int NH) {
    const int lane  = threadIdx.x & 31;
    const int eBase = blockIdx.x << 4;
    const int cols  = NH << 4;
    const int grp   = lane >> 4;
    const int n     = lane & 15;

    // A fragments (16 edge rows x 64) — reused for every head
    v16h a0 = load_a_frag(ef, DIN, eBase, 0,  lane);
    v16h a1 = load_a_frag(ef, DIN, eBase, 32, lane);

    int gs[8], gd[8];
#pragma unroll
    for (int r = 0; r < 8; ++r) {
        int e = eBase + r + (grp << 3);
        gs[r] = src[e];
        gd[r] = dst[e];
    }

    for (int ct = 0; ct < NH; ++ct) {          // one 16-col tile == one head
        FVec acc;
#pragma unroll
        for (int r = 0; r < 8; ++r) acc.f[r] = 0.0f;
        v16h b0 = load_b_frag(W, cols, 0,  ct << 4, lane);
        acc.v = __builtin_amdgcn_wmma_f32_16x16x32_f16(false, a0, false, b0,
                                                       (short)0, acc.v, false, false);
        v16h b1 = load_b_frag(W, cols, 32, ct << 4, lane);
        acc.v = __builtin_amdgcn_wmma_f32_16x16x32_f16(false, a1, false, b1,
                                                       (short)0, acc.v, false, false);

        const int   col = (ct << 4) + n;
        const float av  = att[col];
        float p[8];
#pragma unroll
        for (int r = 0; r < 8; ++r) {
            float v = acc.f[r]
                    + XL[(size_t)gs[r] * cols + col]
                    + XR[(size_t)gd[r] * cols + col];
            v = (v > 0.0f) ? v : 0.2f * v;      // leaky_relu(0.2)
            p[r] = v * av;
        }
        // reduce over the 16 lanes of each half-wave group (wave32, xor masks < 16)
#pragma unroll
        for (int off = 1; off < 16; off <<= 1) {
#pragma unroll
            for (int r = 0; r < 8; ++r)
                p[r] += __shfl_xor(p[r], off, 32);
        }
        if (n == 0) {
#pragma unroll
            for (int r = 0; r < 8; ++r) {
                int e = eBase + r + (grp << 3);
                logits[(size_t)e * NH + ct] = p[r];
                atomicMax(&maxU[(size_t)gd[r] * NH + ct], ordEnc(p[r]));
            }
        }
    }
}

// ---------- softmax pass 2: ex = exp(l - max[dst]); den += ex ----------
__global__ void softmax_exp(const int* __restrict__ dst, float* __restrict__ logits,
                            const unsigned* __restrict__ maxU, float* __restrict__ den,
                            int NH, long long total) {
    long long i = (long long)blockIdx.x * blockDim.x + threadIdx.x;
    if (i >= total) return;
    int e = (int)(i / NH);
    int h = (int)(i % NH);
    int d = dst[e];
    float ex = __expf(logits[i] - ordDec(maxU[(size_t)d * NH + h]));
    logits[i] = ex;
    atomicAdd(&den[(size_t)d * NH + h], ex);
}

// ---------- softmax pass 3: out[dst] += xl[src] * ex/den[dst] ----------
__global__ void aggregate(const int* __restrict__ src, const int* __restrict__ dst,
                          const float* __restrict__ XL, const float* __restrict__ ex,
                          const float* __restrict__ den, float* __restrict__ out,
                          int NH, long long total) {
    long long i = (long long)blockIdx.x * blockDim.x + threadIdx.x;
    if (i >= total) return;
    const int cols = NH << 4;
    int e   = (int)(i / cols);
    int col = (int)(i % cols);
    int h   = col >> 4;
    int s = src[e], d = dst[e];
    float a = ex[(size_t)e * NH + h] / den[(size_t)d * NH + h];
    atomicAdd(&out[(size_t)d * cols + col], XL[(size_t)s * cols + col] * a);
}

// ---------- bias + ELU (in place) ----------
__global__ void bias_elu(float* __restrict__ h, const float* __restrict__ b,
                         int cols, long long total) {
    long long i = (long long)blockIdx.x * blockDim.x + threadIdx.x;
    if (i >= total) return;
    float v = h[i] + b[i % cols];
    h[i] = (v > 0.0f) ? v : expm1f(v);
}

// ---------- finalize: +b2, write h2, write log_softmax(h2) ----------
__global__ void finalize(const float* __restrict__ out2, const float* __restrict__ b2,
                         float* __restrict__ dOut, int Nn) {
    int node = blockIdx.x * blockDim.x + threadIdx.x;
    if (node >= Nn) return;
    float row[DOUT];
    float mx = -1e30f;
#pragma unroll
    for (int c = 0; c < DOUT; ++c) {
        row[c] = out2[(size_t)node * DOUT + c] + b2[c];
        mx = fmaxf(mx, row[c]);
    }
    float s = 0.0f;
#pragma unroll
    for (int c = 0; c < DOUT; ++c) s += __expf(row[c] - mx);
    float lse = mx + __logf(s);
#pragma unroll
    for (int c = 0; c < DOUT; ++c) {
        dOut[(size_t)node * DOUT + c] = row[c];
        dOut[(size_t)Nn * DOUT + (size_t)node * DOUT + c] = row[c] - lse;
    }
}

// ---------- fills ----------
__global__ void fill_f32(float* __restrict__ p, float v, long long n) {
    long long i = (long long)blockIdx.x * blockDim.x + threadIdx.x;
    if (i < n) p[i] = v;
}
__global__ void fill_u32(unsigned* __restrict__ p, unsigned v, long long n) {
    long long i = (long long)blockIdx.x * blockDim.x + threadIdx.x;
    if (i < n) p[i] = v;
}

static inline unsigned host_ord_enc(float f) {
    unsigned u; memcpy(&u, &f, 4);
    return (u & 0x80000000u) ? ~u : (u | 0x80000000u);
}

extern "C" void kernel_launch(void* const* d_in, const int* in_sizes, int n_in,
                              void* d_out, int out_size, void* d_ws, size_t ws_size,
                              hipStream_t stream) {
    const float* x    = (const float*)d_in[0];
    const float* ef   = (const float*)d_in[1];
    const float* Wl1  = (const float*)d_in[2];
    const float* Wr1  = (const float*)d_in[3];
    const float* We1  = (const float*)d_in[4];
    const float* att1 = (const float*)d_in[5];
    const float* b1   = (const float*)d_in[6];
    const float* Wl2  = (const float*)d_in[7];
    const float* Wr2  = (const float*)d_in[8];
    const float* We2  = (const float*)d_in[9];
    const float* att2 = (const float*)d_in[10];
    const float* b2   = (const float*)d_in[11];
    const int*   ei   = (const int*)d_in[12];
    const int* src = ei;
    const int* dst = ei + EE;
    float* dOut = (float*)d_out;

    // workspace carve-up (floats)
    float* ws = (float*)d_ws;
    float*    XL1   = ws;                                  // N*128
    float*    XR1   = XL1 + (size_t)NN * C1;               // N*128
    float*    H1    = XR1 + (size_t)NN * C1;               // N*128 (agg -> h)
    float*    EX1   = H1  + (size_t)NN * C1;               // E*8 (logits -> ex)
    float*    DEN1  = EX1 + (size_t)EE * HEADS;            // N*8
    unsigned* MAX1U = (unsigned*)(DEN1 + (size_t)NN * HEADS); // N*8
    float*    XL2   = (float*)(MAX1U + (size_t)NN * HEADS);   // N*16
    float*    XR2   = XL2 + (size_t)NN * DOUT;             // N*16
    float*    OUT2  = XR2 + (size_t)NN * DOUT;             // N*16
    float*    EX2   = OUT2 + (size_t)NN * DOUT;            // E
    float*    DEN2  = EX2 + (size_t)EE;                    // N
    unsigned* MAX2U = (unsigned*)(DEN2 + (size_t)NN);      // N

    const unsigned NEG_INF_ENC = host_ord_enc(-1e30f);
    const int TB = 256;
    auto blks = [](long long n, int tb) { return (unsigned)((n + tb - 1) / tb); };
    const unsigned gemmBlks = (unsigned)((NN / 16 + 3) / 4);   // 4 waves / block

    // zero / -inf init
    fill_f32<<<blks((long long)NN * C1, TB), TB, 0, stream>>>(H1, 0.0f, (long long)NN * C1);
    fill_f32<<<blks((long long)NN * DOUT, TB), TB, 0, stream>>>(OUT2, 0.0f, (long long)NN * DOUT);
    fill_f32<<<blks((long long)NN * HEADS, TB), TB, 0, stream>>>(DEN1, 0.0f, (long long)NN * HEADS);
    fill_f32<<<blks(NN, TB), TB, 0, stream>>>(DEN2, 0.0f, NN);
    fill_u32<<<blks((long long)NN * HEADS, TB), TB, 0, stream>>>(MAX1U, NEG_INF_ENC, (long long)NN * HEADS);
    fill_u32<<<blks(NN, TB), TB, 0, stream>>>(MAX2U, NEG_INF_ENC, NN);

    // ---- layer 1 ----
    // XL1 = x@Wl1, XR1 = x@Wr1 in one pass (A fragments loaded once)
    wmma_dual_gemm<2, 8><<<gemmBlks, 128, 0, stream>>>(x, Wl1, Wr1, XL1, XR1, NN);
    edge_attn_logits<<<EE / 16, 32, 0, stream>>>(ef, We1, XL1, XR1, att1, src, dst,
                                                 EX1, MAX1U, HEADS);
    softmax_exp<<<blks((long long)EE * HEADS, TB), TB, 0, stream>>>(dst, EX1, MAX1U, DEN1,
                                                                    HEADS, (long long)EE * HEADS);
    aggregate<<<blks((long long)EE * C1, TB), TB, 0, stream>>>(src, dst, XL1, EX1, DEN1, H1,
                                                               HEADS, (long long)EE * C1);
    bias_elu<<<blks((long long)NN * C1, TB), TB, 0, stream>>>(H1, b1, C1, (long long)NN * C1);

    // ---- layer 2 ----
    wmma_dual_gemm<4, 1><<<gemmBlks, 128, 0, stream>>>(H1, Wl2, Wr2, XL2, XR2, NN);
    edge_attn_logits<<<EE / 16, 32, 0, stream>>>(ef, We2, XL2, XR2, att2, src, dst,
                                                 EX2, MAX2U, 1);
    softmax_exp<<<blks(EE, TB), TB, 0, stream>>>(dst, EX2, MAX2U, DEN2, 1, EE);
    aggregate<<<blks((long long)EE * DOUT, TB), TB, 0, stream>>>(src, dst, XL2, EX2, DEN2, OUT2,
                                                                 1, (long long)EE * DOUT);
    finalize<<<blks(NN, TB), TB, 0, stream>>>(OUT2, b2, dOut, NN);
}
